// GPSLayer_22419729285551
// MI455X (gfx1250) — compile-verified, matrix operand order checked by reading, not run on hardware
//
#include <hip/hip_runtime.h>
#include <hip/hip_bf16.h>
#include <cstdint>

#define Nn 4096
#define Dd 1024
#define Hh 8
#define DHd 128
#define Ee 131072
#define EPSf 1e-5f

typedef __attribute__((ext_vector_type(16))) __bf16 v16bf;
typedef __attribute__((ext_vector_type(8)))  float  v8f;

#define WAIT_ASYNC(n) asm volatile("s_wait_asynccnt " #n ::: "memory")

// ---- fp32 -> bf16 round-to-nearest-even ----
__device__ inline unsigned short f2bf(float f) {
    unsigned int u = __float_as_uint(f);
    return (unsigned short)((u + 0x7FFFu + ((u >> 16) & 1u)) >> 16);
}

// ---- CDNA5 async global->LDS 16B copy (ASYNCcnt-tracked, no VGPR staging) ----
__device__ inline void async_cp16(const void* g, const void* l) {
    unsigned long long ga = (unsigned long long)(size_t)g;
    unsigned la = (unsigned)(size_t)l;   // LDS offset lives in low 32 bits of generic ptr
    asm volatile("global_load_async_to_lds_b128 %0, %1, off" :: "v"(la), "v"(ga) : "memory");
}

// Load a 16x32 bf16 A/B-operand fragment from LDS row-major [rows][stride].
// Per CDNA5 ISA layout: lane l holds row (l&15); VGPRs 0-3 hold K = 8*(l>>4)+0..7,
// VGPRs 4-7 hold K = 16 + 8*(l>>4)+0..7  -> two 16B ds_load_b128.
__device__ inline v16bf load_frag(const unsigned short* base, int stride, int row0, int k0) {
    int lane = threadIdx.x & 31;
    const unsigned short* p = base + (size_t)(row0 + (lane & 15)) * stride + k0 + ((lane >> 4) << 3);
    union { v16bf v; uint4 q[2]; } u;
    u.q[0] = *(const uint4*)(p);
    u.q[1] = *(const uint4*)(p + 16);
    return u.v;
}

__device__ inline v8f wmma_bf16(v16bf a, v16bf b, v8f c) {
    return __builtin_amdgcn_wmma_f32_16x16x32_bf16(false, a, false, b, (short)0, c, false, false);
}

// ======================= bf16 GEMM: C[M,N] = A[M,K] @ Bt[N,K]^T (+bias)(+relu) ============
// A, Bt are bf16 in global. 256 threads (8 waves), tile 128x128, BK=32,
// double-buffered LDS filled by async b128 copies. Wave grid 4(M) x 2(N),
// each wave computes 32x64 = 2x4 WMMA tiles.
__global__ __launch_bounds__(256) void gemm_bf16_kernel(
    const unsigned short* __restrict__ A, const unsigned short* __restrict__ Bt,
    const float* __restrict__ bias, void* __restrict__ Cout,
    int M, int K, int N, int relu, int out_bf16)
{
    __shared__ unsigned short As[2][128 * 40];
    __shared__ unsigned short Bs[2][128 * 40];

    const int tid  = threadIdx.x;
    const int lane = tid & 31;
    const int wave = tid >> 5;
    const int wm   = wave & 3;
    const int wn   = wave >> 2;
    const int rowBase = blockIdx.y * 128;
    const int colBase = blockIdx.x * 128;

    // two 16B chunks per thread per tile: chunk c -> row c>>2, k-offset (c&3)*8
    int crow[2], ckc[2];
    #pragma unroll
    for (int it = 0; it < 2; ++it) {
        int c = tid + it * 256;
        crow[it] = c >> 2;
        ckc[it]  = (c & 3) << 3;
    }

    auto prefetch = [&](int kt, int pb) {
        int k0 = kt * 32;
        #pragma unroll
        for (int it = 0; it < 2; ++it)
            async_cp16(A + (size_t)(rowBase + crow[it]) * K + k0 + ckc[it],
                       &As[pb][crow[it] * 40 + ckc[it]]);
        #pragma unroll
        for (int it = 0; it < 2; ++it)
            async_cp16(Bt + (size_t)(colBase + crow[it]) * K + k0 + ckc[it],
                       &Bs[pb][crow[it] * 40 + ckc[it]]);
    };

    v8f acc[2][4];
    #pragma unroll
    for (int i = 0; i < 2; ++i)
        #pragma unroll
        for (int j = 0; j < 4; ++j)
            #pragma unroll
            for (int r = 0; r < 8; ++r) acc[i][j][r] = 0.f;

    const int KT = K >> 5;
    prefetch(0, 0);
    for (int kt = 0; kt < KT; ++kt) {
        int pb = kt & 1;
        if (kt + 1 < KT) {
            prefetch(kt + 1, pb ^ 1);
            WAIT_ASYNC(4);           // tile kt (first 4 in-order async ops) complete
        } else {
            WAIT_ASYNC(0);
        }
        __syncthreads();

        v16bf af[2], bfr[4];
        #pragma unroll
        for (int i = 0; i < 2; ++i) af[i]  = load_frag(As[pb], 40, wm * 32 + i * 16, 0);
        #pragma unroll
        for (int j = 0; j < 4; ++j) bfr[j] = load_frag(Bs[pb], 40, wn * 64 + j * 16, 0);
        #pragma unroll
        for (int i = 0; i < 2; ++i)
            #pragma unroll
            for (int j = 0; j < 4; ++j)
                acc[i][j] = wmma_bf16(af[i], bfr[j], acc[i][j]);
        __syncthreads();
    }

    // Epilogue. bias always valid (zero buffer when unused); hoisted loads.
    float bv[4];
    #pragma unroll
    for (int j = 0; j < 4; ++j)
        bv[j] = bias[colBase + wn * 64 + j * 16 + (lane & 15)];

    if (out_bf16) {
        unsigned short* C = (unsigned short*)Cout;
        #pragma unroll
        for (int i = 0; i < 2; ++i)
            #pragma unroll
            for (int j = 0; j < 4; ++j)
                #pragma unroll
                for (int r = 0; r < 8; ++r) {
                    int row = rowBase + wm * 32 + i * 16 + r + ((lane >> 4) << 3);
                    int col = colBase + wn * 64 + j * 16 + (lane & 15);
                    float v = acc[i][j][r] + bv[j];
                    if (relu) v = fmaxf(v, 0.f);
                    C[(size_t)row * N + col] = f2bf(v);
                }
    } else {
        float* C = (float*)Cout;
        #pragma unroll
        for (int i = 0; i < 2; ++i)
            #pragma unroll
            for (int j = 0; j < 4; ++j)
                #pragma unroll
                for (int r = 0; r < 8; ++r) {
                    int row = rowBase + wm * 32 + i * 16 + r + ((lane >> 4) << 3);
                    int col = colBase + wn * 64 + j * 16 + (lane & 15);
                    float v = acc[i][j][r] + bv[j];
                    if (relu) v = fmaxf(v, 0.f);
                    C[(size_t)row * N + col] = v;
                }
    }
}

// ======================= Flash attention (bf16 in / bf16 out) ==================
// grid (N/128, H), block 256 = 8 waves; each wave owns 16 query rows.
__global__ __launch_bounds__(256) void attn_kernel(
    const unsigned short* __restrict__ Q, const unsigned short* __restrict__ Km,
    const unsigned short* __restrict__ Vm, unsigned short* __restrict__ O)
{
    __shared__ unsigned short Qs[128 * 136];      // [qrow][dh]
    __shared__ unsigned short Ks[32 * 136];       // [key][dh]
    __shared__ unsigned short Vt[128 * 40];       // [dh][key]
    __shared__ unsigned short Ps[8 * 16 * 40];    // per-wave P staging

    const int tid  = threadIdx.x;
    const int lane = tid & 31;
    const int wave = tid >> 5;
    const int qb   = blockIdx.x * 128;
    const int hb   = blockIdx.y;
    const float scale = 0.08838834764831845f;     // 1/sqrt(128)

    // stage Q block via async copies (8 x 16B per thread)
    #pragma unroll
    for (int it = 0; it < 8; ++it) {
        int c = tid + it * 256;
        int r = c >> 4, kc = (c & 15) << 3;
        async_cp16(Q + (size_t)(qb + r) * Dd + hb * DHd + kc, &Qs[r * 136 + kc]);
    }

    v8f acc[8];
    #pragma unroll
    for (int dI = 0; dI < 8; ++dI)
        #pragma unroll
        for (int r = 0; r < 8; ++r) acc[dI][r] = 0.f;
    float mrow[8], lrow[8];
    #pragma unroll
    for (int r = 0; r < 8; ++r) { mrow[r] = -3.0e38f; lrow[r] = 0.f; }

    for (int kb = 0; kb < Nn; kb += 32) {
        __syncthreads();
        // K tile: async straight copy; V tile: manual transposed store
        #pragma unroll
        for (int it = 0; it < 2; ++it) {
            int c = tid + it * 256;
            int r = c >> 4, kc = (c & 15) << 3;
            async_cp16(Km + (size_t)(kb + r) * Dd + hb * DHd + kc, &Ks[r * 136 + kc]);
        }
        #pragma unroll
        for (int it = 0; it < 2; ++it) {
            int c = tid + it * 256;
            int r = c >> 4, c8 = (c & 15) << 3;
            uint4 dv = *(const uint4*)(Vm + (size_t)(kb + r) * Dd + hb * DHd + c8);
            Vt[(c8 + 0) * 40 + r] = (unsigned short)(dv.x);
            Vt[(c8 + 1) * 40 + r] = (unsigned short)(dv.x >> 16);
            Vt[(c8 + 2) * 40 + r] = (unsigned short)(dv.y);
            Vt[(c8 + 3) * 40 + r] = (unsigned short)(dv.y >> 16);
            Vt[(c8 + 4) * 40 + r] = (unsigned short)(dv.z);
            Vt[(c8 + 5) * 40 + r] = (unsigned short)(dv.z >> 16);
            Vt[(c8 + 6) * 40 + r] = (unsigned short)(dv.w);
            Vt[(c8 + 7) * 40 + r] = (unsigned short)(dv.w >> 16);
        }
        WAIT_ASYNC(0);
        __syncthreads();

        // S = Q K^T : 2 key-tiles x 4 dh-steps per wave
        v8f s[2];
        #pragma unroll
        for (int n = 0; n < 2; ++n)
            #pragma unroll
            for (int r = 0; r < 8; ++r) s[n][r] = 0.f;
        #pragma unroll
        for (int ks = 0; ks < 4; ++ks) {
            v16bf aq = load_frag(Qs, 136, wave * 16, ks * 32);
            #pragma unroll
            for (int n = 0; n < 2; ++n) {
                v16bf bk = load_frag(Ks, 136, n * 16, ks * 32);
                s[n] = wmma_bf16(aq, bk, s[n]);
            }
        }
        #pragma unroll
        for (int n = 0; n < 2; ++n)
            #pragma unroll
            for (int r = 0; r < 8; ++r) s[n][r] *= scale;

        // online softmax: row lives in one VGPR across a 16-lane half-wave
        float alpha[8];
        #pragma unroll
        for (int r = 0; r < 8; ++r) {
            float mx = fmaxf(s[0][r], s[1][r]);
            mx = fmaxf(mx, __shfl_xor(mx, 1, 32));
            mx = fmaxf(mx, __shfl_xor(mx, 2, 32));
            mx = fmaxf(mx, __shfl_xor(mx, 4, 32));
            mx = fmaxf(mx, __shfl_xor(mx, 8, 32));
            float mn = fmaxf(mrow[r], mx);
            alpha[r] = __expf(mrow[r] - mn);
            mrow[r]  = mn;
        }
        #pragma unroll
        for (int r = 0; r < 8; ++r) {
            float rs = 0.f;
            #pragma unroll
            for (int n = 0; n < 2; ++n) {
                float p = __expf(s[n][r] - mrow[r]);
                s[n][r] = p;
                rs += p;
            }
            rs += __shfl_xor(rs, 1, 32);
            rs += __shfl_xor(rs, 2, 32);
            rs += __shfl_xor(rs, 4, 32);
            rs += __shfl_xor(rs, 8, 32);
            lrow[r] = lrow[r] * alpha[r] + rs;
            #pragma unroll
            for (int dI = 0; dI < 8; ++dI) acc[dI][r] *= alpha[r];
        }

        // stage P (C-layout -> row-major LDS) as next A-operand
        {
            int mh = ((lane >> 4) << 3);
            int nc = lane & 15;
            #pragma unroll
            for (int n = 0; n < 2; ++n)
                #pragma unroll
                for (int r = 0; r < 8; ++r)
                    Ps[(size_t)(wave * 16 + r + mh) * 40 + n * 16 + nc] = f2bf(s[n][r]);
        }
        asm volatile("s_wait_dscnt 0" ::: "memory");

        // O += P @ V
        v16bf ap = load_frag(Ps, 40, wave * 16, 0);
        #pragma unroll
        for (int dI = 0; dI < 8; ++dI) {
            v16bf bv = load_frag(Vt, 40, dI * 16, 0);
            acc[dI] = wmma_bf16(ap, bv, acc[dI]);
        }
    }

    #pragma unroll
    for (int dI = 0; dI < 8; ++dI)
        #pragma unroll
        for (int r = 0; r < 8; ++r) {
            int row = qb + wave * 16 + r + ((lane >> 4) << 3);
            int col = hb * DHd + dI * 16 + (lane & 15);
            O[(size_t)row * Dd + col] = f2bf(acc[dI][r] / lrow[r]);
        }
}

// ======================= conversion / transpose ==================
__global__ void cvt_bf16_kernel(const float* __restrict__ in, unsigned short* __restrict__ out) {
    size_t i = (size_t)blockIdx.x * blockDim.x + threadIdx.x;
    out[i] = f2bf(in[i]);
}
// out[n][k] = bf16(in[k][n]); in fp32 [K][N]
__global__ void transpose_cvt_kernel(const float* __restrict__ in,
                                     unsigned short* __restrict__ out, int K, int N) {
    __shared__ float t[32][33];
    int kb = blockIdx.y * 32, nb = blockIdx.x * 32;
    int tx = threadIdx.x & 31, ty = threadIdx.x >> 5;   // 256 threads
    #pragma unroll
    for (int j = 0; j < 32; j += 8)
        t[ty + j][tx] = in[(size_t)(kb + ty + j) * N + nb + tx];
    __syncthreads();
    #pragma unroll
    for (int j = 0; j < 32; j += 8)
        out[(size_t)(nb + ty + j) * K + kb + tx] = f2bf(t[tx][ty + j]);
}

// ======================= GCN scatter ==================
__global__ void deg_init_kernel(float* deg) {
    int i = blockIdx.x * blockDim.x + threadIdx.x;
    if (i < Nn) deg[i] = 1.0f;                      // self-loop
}
__global__ void deg_count_kernel(const int* __restrict__ ei, float* deg) {
    int e = blockIdx.x * blockDim.x + threadIdx.x;
    if (e < Ee) atomicAdd(&deg[ei[Ee + e]], 1.0f);
}
__global__ void gcn_self_kernel(const float* __restrict__ hlin, const float* __restrict__ deg,
                                const float* __restrict__ bias, float* __restrict__ out) {
    int row = blockIdx.x;
    float inv = 1.0f / deg[row];
    #pragma unroll
    for (int j = 0; j < 4; ++j) {
        int c = threadIdx.x + j * 256;
        out[(size_t)row * Dd + c] = bias[c] + hlin[(size_t)row * Dd + c] * inv;
    }
}
__global__ void gcn_edge_kernel(const int* __restrict__ ei, const float* __restrict__ hlin,
                                const float* __restrict__ deg, float* __restrict__ out) {
    int e = blockIdx.x;
    int src = ei[e], dst = ei[Ee + e];
    float norm = rsqrtf(deg[src]) * rsqrtf(deg[dst]);
    #pragma unroll
    for (int j = 0; j < 4; ++j) {
        int c = threadIdx.x + j * 256;
        atomicAdd(&out[(size_t)dst * Dd + c], hlin[(size_t)src * Dd + c] * norm);
    }
}

// ======================= BatchNorm ==================
__global__ void bn_stats_kernel(const float* __restrict__ a, const float* __restrict__ b,
                                float* __restrict__ sum, float* __restrict__ sumsq) {
    int c0 = threadIdx.x;
    float s[4] = {0, 0, 0, 0}, q[4] = {0, 0, 0, 0};
    int r0 = blockIdx.x * 64;
    for (int rr = 0; rr < 64; ++rr) {
        size_t base = (size_t)(r0 + rr) * Dd;
        #pragma unroll
        for (int j = 0; j < 4; ++j) {
            float h = a[base + c0 + j * 256] + b[base + c0 + j * 256];
            s[j] += h; q[j] += h * h;
        }
    }
    #pragma unroll
    for (int j = 0; j < 4; ++j) {
        atomicAdd(&sum[c0 + j * 256], s[j]);
        atomicAdd(&sumsq[c0 + j * 256], q[j]);
    }
}
__global__ void bn_apply_kernel(const float* __restrict__ a, const float* __restrict__ b,
                                const float* __restrict__ sum, const float* __restrict__ sumsq,
                                const float* __restrict__ g, const float* __restrict__ beta,
                                float* __restrict__ out) {
    size_t i = (size_t)blockIdx.x * blockDim.x + threadIdx.x;
    int c = (int)(i & (Dd - 1));
    float mu  = sum[c] * (1.0f / Nn);
    float var = sumsq[c] * (1.0f / Nn) - mu * mu;
    float h = a[i] + b[i];
    out[i] = (h - mu) * rsqrtf(var + EPSf) * g[c] + beta[c];
}
__global__ void add_dual_kernel(const float* __restrict__ a, const float* __restrict__ b,
                                float* __restrict__ out, unsigned short* __restrict__ outb) {
    size_t i = (size_t)blockIdx.x * blockDim.x + threadIdx.x;
    float v = a[i] + b[i];
    out[i]  = v;
    outb[i] = f2bf(v);
}

// ======================= driver ==================
extern "C" void kernel_launch(void* const* d_in, const int* in_sizes, int n_in,
                              void* d_out, int out_size, void* d_ws, size_t ws_size,
                              hipStream_t stream)
{
    (void)in_sizes; (void)n_in; (void)out_size; (void)ws_size;

    const float* x      = (const float*)d_in[0];
    const int*   ei     = (const int*)d_in[1];
    const float* gcn_w  = (const float*)d_in[2];
    const float* gcn_b  = (const float*)d_in[3];
    const float* wq = (const float*)d_in[4];  const float* bq = (const float*)d_in[5];
    const float* wk = (const float*)d_in[6];  const float* bk = (const float*)d_in[7];
    const float* wv = (const float*)d_in[8];  const float* bv = (const float*)d_in[9];
    const float* wo = (const float*)d_in[10]; const float* bo = (const float*)d_in[11];
    const float* bn1l_g = (const float*)d_in[12]; const float* bn1l_b = (const float*)d_in[13];
    const float* bn1a_g = (const float*)d_in[14]; const float* bn1a_b = (const float*)d_in[15];
    const float* bn2_g  = (const float*)d_in[16]; const float* bn2_b  = (const float*)d_in[17];
    const float* ff_w1 = (const float*)d_in[18]; const float* ff_b1 = (const float*)d_in[19];
    const float* ff_w2 = (const float*)d_in[20]; const float* ff_b2 = (const float*)d_in[21];

    const size_t ND = (size_t)Nn * Dd;
    char* wsp = (char*)d_ws;
    auto alloc = [&](size_t bytes) {
        char* p = wsp;
        wsp += (bytes + 255) & ~(size_t)255;
        return p;
    };
    float* F1 = (float*)alloc(ND * 4);
    float* F2 = (float*)alloc(ND * 4);
    float* F3 = (float*)alloc(ND * 4);
    float* F4 = (float*)alloc(ND * 4);
    unsigned short* xb  = (unsigned short*)alloc(ND * 2);
    unsigned short* qb2 = (unsigned short*)alloc(ND * 2);
    unsigned short* kb2 = (unsigned short*)alloc(ND * 2);
    unsigned short* vb2 = (unsigned short*)alloc(ND * 2);
    unsigned short* ob  = (unsigned short*)alloc(ND * 2);
    unsigned short* hb  = (unsigned short*)alloc(ND * 2);
    unsigned short* f1b = (unsigned short*)alloc(ND * 4);          // [N][2D] bf16
    unsigned short* wgT = (unsigned short*)alloc((size_t)Dd * Dd * 2);
    unsigned short* wqT = (unsigned short*)alloc((size_t)Dd * Dd * 2);
    unsigned short* wkT = (unsigned short*)alloc((size_t)Dd * Dd * 2);
    unsigned short* wvT = (unsigned short*)alloc((size_t)Dd * Dd * 2);
    unsigned short* woT = (unsigned short*)alloc((size_t)Dd * Dd * 2);
    unsigned short* w1T = (unsigned short*)alloc((size_t)Dd * 2 * Dd * 2);
    unsigned short* w2T = (unsigned short*)alloc((size_t)Dd * 2 * Dd * 2);
    float* zbias = (float*)alloc(2 * Dd * 4);
    float* deg   = (float*)alloc(Nn * 4);
    float* bsum  = (float*)alloc(Dd * 4);
    float* bsq   = (float*)alloc(Dd * 4);

    dim3 blk(256);
    dim3 g1(Dd / 128, Nn / 128);
    dim3 g2(2 * Dd / 128, Nn / 128);

    hipMemsetAsync(zbias, 0, 2 * Dd * sizeof(float), stream);

    // ---- one-time bf16 conversions ----
    cvt_bf16_kernel<<<ND / 256, blk, 0, stream>>>(x, xb);
    transpose_cvt_kernel<<<dim3(Dd / 32, Dd / 32), blk, 0, stream>>>(gcn_w, wgT, Dd, Dd);
    transpose_cvt_kernel<<<dim3(Dd / 32, Dd / 32), blk, 0, stream>>>(wq, wqT, Dd, Dd);
    transpose_cvt_kernel<<<dim3(Dd / 32, Dd / 32), blk, 0, stream>>>(wk, wkT, Dd, Dd);
    transpose_cvt_kernel<<<dim3(Dd / 32, Dd / 32), blk, 0, stream>>>(wv, wvT, Dd, Dd);
    transpose_cvt_kernel<<<dim3(Dd / 32, Dd / 32), blk, 0, stream>>>(wo, woT, Dd, Dd);
    transpose_cvt_kernel<<<dim3(2 * Dd / 32, Dd / 32), blk, 0, stream>>>(ff_w1, w1T, Dd, 2 * Dd);
    transpose_cvt_kernel<<<dim3(Dd / 32, 2 * Dd / 32), blk, 0, stream>>>(ff_w2, w2T, 2 * Dd, Dd);

    // ---- GCN branch ----
    gemm_bf16_kernel<<<g1, blk, 0, stream>>>(xb, wgT, zbias, F1, Nn, Dd, Dd, 0, 0);
    deg_init_kernel<<<Nn / 256, blk, 0, stream>>>(deg);
    deg_count_kernel<<<Ee / 256, blk, 0, stream>>>(ei, deg);
    gcn_self_kernel<<<Nn, blk, 0, stream>>>(F1, deg, gcn_b, F2);
    gcn_edge_kernel<<<Ee, blk, 0, stream>>>(ei, F1, deg, F2);
    hipMemsetAsync(bsum, 0, 2 * Dd * sizeof(float), stream);
    bn_stats_kernel<<<Nn / 64, blk, 0, stream>>>(x, F2, bsum, bsq);
    bn_apply_kernel<<<ND / 256, blk, 0, stream>>>(x, F2, bsum, bsq, bn1l_g, bn1l_b, F1);
    // F1 = h_local

    // ---- attention branch (bf16 end-to-end) ----
    gemm_bf16_kernel<<<g1, blk, 0, stream>>>(xb, wqT, bq, qb2, Nn, Dd, Dd, 0, 1);
    gemm_bf16_kernel<<<g1, blk, 0, stream>>>(xb, wkT, bk, kb2, Nn, Dd, Dd, 0, 1);
    gemm_bf16_kernel<<<g1, blk, 0, stream>>>(xb, wvT, bv, vb2, Nn, Dd, Dd, 0, 1);
    attn_kernel<<<dim3(Nn / 128, Hh), blk, 0, stream>>>(qb2, kb2, vb2, ob);
    gemm_bf16_kernel<<<g1, blk, 0, stream>>>(ob, woT, bo, F2, Nn, Dd, Dd, 0, 0);
    hipMemsetAsync(bsum, 0, 2 * Dd * sizeof(float), stream);
    bn_stats_kernel<<<Nn / 64, blk, 0, stream>>>(x, F2, bsum, bsq);
    bn_apply_kernel<<<ND / 256, blk, 0, stream>>>(x, F2, bsum, bsq, bn1a_g, bn1a_b, F3);
    // F3 = h_attn

    // ---- combine + FFN ----
    add_dual_kernel<<<ND / 256, blk, 0, stream>>>(F1, F3, F4, hb);   // h (fp32 + bf16)
    gemm_bf16_kernel<<<g2, blk, 0, stream>>>(hb, w1T, ff_b1, f1b, Nn, Dd, 2 * Dd, 1, 1);
    gemm_bf16_kernel<<<g1, blk, 0, stream>>>(f1b, w2T, ff_b2, F1, Nn, 2 * Dd, Dd, 0, 0);
    hipMemsetAsync(bsum, 0, 2 * Dd * sizeof(float), stream);
    bn_stats_kernel<<<Nn / 64, blk, 0, stream>>>(F4, F1, bsum, bsq);
    bn_apply_kernel<<<ND / 256, blk, 0, stream>>>(F4, F1, bsum, bsq, bn2_g, bn2_b,
                                                  (float*)d_out);
}
